// Net2_20993800143103
// MI455X (gfx1250) — compile-verified, matrix operand order checked by reading, not run on hardware
//
#include <hip/hip_runtime.h>
#include <hip/hip_bf16.h>

#define N_NODES 100000
#define N_EDGES 3200000
#define N_GRAPHS 128
#define D_IN 128
#define HID 64
#define D_OUT 32

typedef __attribute__((ext_vector_type(16))) __bf16 v16bf;
typedef __attribute__((ext_vector_type(8)))  float v8f;

static __device__ __forceinline__ __bf16 f2bf(float f) {
    unsigned int u = __builtin_bit_cast(unsigned int, f);
    unsigned short h = (unsigned short)(u >> 16);   // truncate to bf16
    return __builtin_bit_cast(__bf16, h);
}

// ---------------------------------------------------------------------------
// Dual-output GEMM: P = A @ Wrel  (no bias),  G = A @ Wroot + bias
// A:[N,K] f32 row-major, Wrel/Wroot:[K,64] f32 row-major, bias:[64]
// One wave per 16x16 output tile; block = 4 waves = one row-tile x 4 col-tiles.
// bf16 WMMA 16x16x32, f32 accumulate. N % 16 == 0 (EXEC all-ones for WMMA).
// ---------------------------------------------------------------------------
__global__ void gemm_dual_kernel(const float* __restrict__ A,
                                 const float* __restrict__ Wrel,
                                 const float* __restrict__ Wroot,
                                 const float* __restrict__ bias,
                                 float* __restrict__ P,
                                 float* __restrict__ G,
                                 int K) {
    const int wave    = threadIdx.x >> 5;      // 0..3 -> column tile
    const int lane    = threadIdx.x & 31;
    const int rowBase = blockIdx.x * 16;
    const int colBase = wave * 16;

    const int r     = lane & 15;               // A row within tile / D column
    const int hi    = lane >> 4;               // half-wave select
    const int kbase = hi * 8;                  // K sub-offset per ISA layout

    const float* __restrict__ arow = A + (size_t)(rowBase + r) * K;
    const int c = colBase + r;                 // B/D column for this lane

    v8f accP = {0.f,0.f,0.f,0.f,0.f,0.f,0.f,0.f};
    v8f accG = {0.f,0.f,0.f,0.f,0.f,0.f,0.f,0.f};

    for (int kk = 0; kk < K; kk += 32) {
        // ---- A fragment (16x32 bf16): two contiguous runs of 8 f32 per lane
        const float4 a0 = *(const float4*)(arow + kk + kbase);
        const float4 a1 = *(const float4*)(arow + kk + kbase + 4);
        const float4 a2 = *(const float4*)(arow + kk + kbase + 16);
        const float4 a3 = *(const float4*)(arow + kk + kbase + 20);
        v16bf af;
        af[0]=f2bf(a0.x); af[1]=f2bf(a0.y); af[2]=f2bf(a0.z); af[3]=f2bf(a0.w);
        af[4]=f2bf(a1.x); af[5]=f2bf(a1.y); af[6]=f2bf(a1.z); af[7]=f2bf(a1.w);
        af[8]=f2bf(a2.x); af[9]=f2bf(a2.y); af[10]=f2bf(a2.z); af[11]=f2bf(a2.w);
        af[12]=f2bf(a3.x); af[13]=f2bf(a3.y); af[14]=f2bf(a3.z); af[15]=f2bf(a3.w);

        // ---- B fragments (32x16 bf16), column c, k strided by 64
        const float* __restrict__ wr = Wrel  + (size_t)(kk + kbase) * HID + c;
        const float* __restrict__ wo = Wroot + (size_t)(kk + kbase) * HID + c;
        v16bf br, bo;
#pragma unroll
        for (int e = 0; e < 8; ++e) {
            br[e]     = f2bf(wr[e * HID]);
            br[e + 8] = f2bf(wr[(16 + e) * HID]);
            bo[e]     = f2bf(wo[e * HID]);
            bo[e + 8] = f2bf(wo[(16 + e) * HID]);
        }

        accP = __builtin_amdgcn_wmma_f32_16x16x32_bf16(
            false, af, false, br, (short)0, accP, false, false);
        accG = __builtin_amdgcn_wmma_f32_16x16x32_bf16(
            false, af, false, bo, (short)0, accG, false, false);
    }

    // ---- Epilogue: D layout -> row M = i + hi*8, col = c
    const float bv = bias[c];
    size_t base = ((size_t)rowBase + hi * 8) * HID + c;
#pragma unroll
    for (int i = 0; i < 8; ++i) {
        P[base + (size_t)i * HID] = accP[i];
        G[base + (size_t)i * HID] = accG[i] + bv;
    }
}

// ---------------------------------------------------------------------------
// Edge scatter: Agg[dst] += P[src].  32 lanes per edge, float2 per lane.
// Random traffic stays in the 192MB L2 (node buffers are 25.6MB).
// ---------------------------------------------------------------------------
__global__ void edge_scatter_kernel(const int* __restrict__ src,
                                    const int* __restrict__ dst,
                                    const float* __restrict__ P,
                                    float* __restrict__ Agg) {
    long long t = (long long)blockIdx.x * blockDim.x + threadIdx.x;
    int e    = (int)(t >> 5);
    int lane = (int)(t & 31);
    if (e >= N_EDGES) return;
    int s = src[e];
    int d = dst[e];
    const float2 v = *(const float2*)(P + (size_t)s * HID + lane * 2);
    float* out = Agg + (size_t)d * HID + lane * 2;
    atomicAdd(out,     v.x);
    atomicAdd(out + 1, v.y);
}

// ---------------------------------------------------------------------------
// F = relu?(Agg)   (Agg already holds agg + root-proj + bias)
// ---------------------------------------------------------------------------
__global__ void combine_kernel(const float* __restrict__ Agg,
                               float* __restrict__ F,
                               int n, int doRelu) {
    int i = blockIdx.x * blockDim.x + threadIdx.x;
    if (i >= n) return;
    float v = Agg[i];
    F[i] = doRelu ? fmaxf(v, 0.f) : v;
}

__global__ void zero_kernel(float* __restrict__ p, int n) {
    int i = blockIdx.x * blockDim.x + threadIdx.x;
    if (i < n) p[i] = 0.f;
}

// ---------------------------------------------------------------------------
// Pool accumulate: sums[batch[node]] += F[node]; cnts[batch[node]] += 1
// ---------------------------------------------------------------------------
__global__ void pool_accum_kernel(const float* __restrict__ F,
                                  const int* __restrict__ batch,
                                  float* __restrict__ sums,
                                  float* __restrict__ cnts) {
    long long t = (long long)blockIdx.x * blockDim.x + threadIdx.x;
    int node = (int)(t >> 5);
    int lane = (int)(t & 31);
    if (node >= N_NODES) return;
    int g = batch[node];
    const float2 v = *(const float2*)(F + (size_t)node * HID + lane * 2);
    float* out = sums + (size_t)g * HID + lane * 2;
    atomicAdd(out,     v.x);
    atomicAdd(out + 1, v.y);
    if (lane == 0) atomicAdd(cnts + g, 1.0f);
}

// ---------------------------------------------------------------------------
// out[g,o] = (sums[g]/max(cnt,1)) @ Wlin + blin   (128x64x32, trivial)
// ---------------------------------------------------------------------------
__global__ void final_linear_kernel(const float* __restrict__ sums,
                                    const float* __restrict__ cnts,
                                    const float* __restrict__ Wlin,
                                    const float* __restrict__ blin,
                                    float* __restrict__ out) {
    int t = blockIdx.x * blockDim.x + threadIdx.x;   // 0..4095
    if (t >= N_GRAPHS * D_OUT) return;
    int g = t >> 5;        // D_OUT == 32
    int o = t & 31;
    float inv = 1.0f / fmaxf(cnts[g], 1.0f);
    float acc = blin[o];
    const float* srow = sums + (size_t)g * HID;
#pragma unroll 8
    for (int k = 0; k < HID; ++k)
        acc += (srow[k] * inv) * Wlin[k * D_OUT + o];
    out[t] = acc;
}

extern "C" void kernel_launch(void* const* d_in, const int* in_sizes, int n_in,
                              void* d_out, int out_size, void* d_ws, size_t ws_size,
                              hipStream_t stream) {
    const float* x       = (const float*)d_in[0];
    const int*   ei      = (const int*)d_in[1];
    const int*   src     = ei;
    const int*   dst     = ei + N_EDGES;
    // d_in[2] = edge_attr : unused by the reference
    const int*   batch   = (const int*)d_in[3];
    const float* W1_rel  = (const float*)d_in[4];
    const float* b1_rel  = (const float*)d_in[5];
    const float* W1_root = (const float*)d_in[6];
    const float* W2_rel  = (const float*)d_in[7];
    const float* b2_rel  = (const float*)d_in[8];
    const float* W2_root = (const float*)d_in[9];
    const float* W3_rel  = (const float*)d_in[10];
    const float* b3_rel  = (const float*)d_in[11];
    const float* W3_root = (const float*)d_in[12];
    const float* W_lin   = (const float*)d_in[13];
    const float* b_lin   = (const float*)d_in[14];

    const size_t NF = (size_t)N_NODES * HID;    // 6.4M floats
    float* P    = (float*)d_ws;                 // projected source features
    float* Agg  = P + NF;                       // aggregation (init = root proj + bias)
    float* F    = Agg + NF;                     // layer activations
    float* sums = F + NF;                       // [128,64]
    float* cnts = sums + (size_t)N_GRAPHS * HID;// [128]

    const int gemmGrid  = N_NODES / 16;                         // 6250 (exact)
    const int edgeGrid  = (int)(((long long)N_EDGES * 32 + 255) / 256);
    const int combGrid  = (int)((NF + 255) / 256);
    const int poolGrid  = (int)(((long long)N_NODES * 32 + 255) / 256);

    // ---- Layer 1 (K = 128), project-then-aggregate
    gemm_dual_kernel<<<gemmGrid, 128, 0, stream>>>(x, W1_rel, W1_root, b1_rel, P, Agg, D_IN);
    edge_scatter_kernel<<<edgeGrid, 256, 0, stream>>>(src, dst, P, Agg);
    combine_kernel<<<combGrid, 256, 0, stream>>>(Agg, F, (int)NF, 1);

    // ---- Layer 2 (K = 64)
    gemm_dual_kernel<<<gemmGrid, 128, 0, stream>>>(F, W2_rel, W2_root, b2_rel, P, Agg, HID);
    edge_scatter_kernel<<<edgeGrid, 256, 0, stream>>>(src, dst, P, Agg);
    combine_kernel<<<combGrid, 256, 0, stream>>>(Agg, F, (int)NF, 1);

    // ---- Layer 3 (K = 64, no relu)
    gemm_dual_kernel<<<gemmGrid, 128, 0, stream>>>(F, W3_rel, W3_root, b3_rel, P, Agg, HID);
    edge_scatter_kernel<<<edgeGrid, 256, 0, stream>>>(src, dst, P, Agg);
    combine_kernel<<<combGrid, 256, 0, stream>>>(Agg, F, (int)NF, 0);

    // ---- Global mean pool + final linear
    zero_kernel<<<(N_GRAPHS * HID + N_GRAPHS + 255) / 256, 256, 0, stream>>>(
        sums, N_GRAPHS * HID + N_GRAPHS);
    pool_accum_kernel<<<poolGrid, 256, 0, stream>>>(F, batch, sums, cnts);
    final_linear_kernel<<<(N_GRAPHS * D_OUT + 255) / 256, 256, 0, stream>>>(
        sums, cnts, W_lin, b_lin, (float*)d_out);
}
